// MultiHeadAttention_20040317403658
// MI455X (gfx1250) — compile-verified, matrix operand order checked by reading, not run on hardware
//
#include <hip/hip_runtime.h>
#include <hip/hip_bf16.h>
#include <math.h>

#define B_  2
#define S_  2048
#define E_  1024
#define H_  16
#define DK_ 64
#define TK  64   // K-tile per GEMM stage (2 WMMA k-slices)
#define LDK 72   // padded LDS row stride (elements); 72*2B=144B, multiple of 16B

typedef __attribute__((ext_vector_type(16))) __bf16 v16bf;
typedef __attribute__((ext_vector_type(8)))  float  v8f;
typedef __attribute__((ext_vector_type(4)))  int    v4i;

union V16U { uint4 u[2]; v16bf v; };

// 32-byte fragment load through two 16B-aligned b128 accesses
static __device__ __forceinline__ v16bf ld16(const __bf16* p) {
  V16U t;
  const uint4* q = reinterpret_cast<const uint4*>(p);
  t.u[0] = q[0];
  t.u[1] = q[1];
  return t.v;
}

static __device__ __forceinline__ v8f wmma_bf16(v16bf a, v16bf b, v8f c) {
  return __builtin_amdgcn_wmma_f32_16x16x32_bf16(
      /*neg_a=*/false, a, /*neg_b=*/false, b,
      /*c_mod=*/(short)0, c, /*reuse_a=*/false, /*reuse_b=*/false);
}

// ---- CDNA5 async global->LDS copy (ASYNCcnt-tracked DMA), with fallback ----
#if __has_builtin(__builtin_amdgcn_global_load_async_to_lds_b128)
#define ASYNC_LDS 1
#else
#define ASYNC_LDS 0
#endif

static __device__ __forceinline__ void cp16_g2l(const __bf16* g, __bf16* l) {
#if ASYNC_LDS
  __builtin_amdgcn_global_load_async_to_lds_b128(
      (__attribute__((address_space(1))) v4i*)(g),
      (__attribute__((address_space(3))) v4i*)(l), 0, 0);
#else
  *reinterpret_cast<uint4*>(l) = *reinterpret_cast<const uint4*>(g);
#endif
}

static __device__ __forceinline__ void cp_wait() {
#if ASYNC_LDS && __has_builtin(__builtin_amdgcn_s_wait_asynccnt)
  __builtin_amdgcn_s_wait_asynccnt(0);
#endif
}

// ---------------- weight fp32 -> bf16 (wq|wk|wv|wo packed) ----------------
__global__ __launch_bounds__(256)
void cvt_w_kernel(const float* __restrict__ wq, const float* __restrict__ wk,
                  const float* __restrict__ wv, const float* __restrict__ wo,
                  __bf16* __restrict__ dst) {
  const int n = E_ * E_;
  int i = blockIdx.x * 256 + threadIdx.x;          // 0 .. 4n-1
  const float* src = (i < n) ? wq : (i < 2 * n) ? wk : (i < 3 * n) ? wv : wo;
  dst[i] = (__bf16)src[i & (n - 1)];
}

// ---------------- fused QKV projection: out = X @ W^T + b ----------------
// Writes Q,K as [B,H,S,DK] bf16 and V transposed as [B,H,DK,S] bf16.
__global__ __launch_bounds__(256)
void gemm_qkv_kernel(const float* __restrict__ xq, const float* __restrict__ xk,
                     const float* __restrict__ xv, const __bf16* __restrict__ wbf,
                     const float* __restrict__ bq, const float* __restrict__ bk,
                     const float* __restrict__ bv,
                     __bf16* __restrict__ Qb, __bf16* __restrict__ Kb,
                     __bf16* __restrict__ Vt) {
  __shared__ __align__(32) __bf16 As[128 * LDK];
  __shared__ __align__(32) __bf16 Bs[128 * LDK];

  const int z = blockIdx.z;
  const float*  X    = (z == 0) ? xq : (z == 1) ? xk : xv;
  const __bf16* W    = wbf + (size_t)z * E_ * E_;
  const float*  bias = (z == 0) ? bq : (z == 1) ? bk : bv;

  const int n0 = blockIdx.x * 128;
  const int m0 = blockIdx.y * 128;
  const int t    = threadIdx.x;
  const int lane = t & 31;
  const int wid  = t >> 5;
  const int wm = (wid & 1) * 64;       // 2 waves along M (64 rows each)
  const int wn = (wid >> 1) * 32;      // 4 waves along N (32 cols each)
  const int fr = lane & 15;            // fragment row (M or N index mod 16)
  const int kh = (lane >> 4) * 16;     // K-half offset within a 32-K fragment
  const int rbase = (lane >> 4) * 8;   // C-fragment: M = r + 8*(lane>>4)

  v8f acc[4][2] = {};

  const int arow = t >> 4, acol = (t & 15) * 4;  // A staging: fp32, 16 thr/row
  const int brow = t >> 3, bcol = (t & 7) * 8;   // B staging: bf16, 8 thr/row

  for (int k0 = 0; k0 < E_; k0 += TK) {
    // stage B tile (128x64 bf16) via async DMA to LDS
    for (int p = 0; p < 4; ++p) {
      int r = brow + p * 32;
      cp16_g2l(W + (size_t)(n0 + r) * E_ + k0 + bcol, &Bs[r * LDK + bcol]);
    }
    // stage A tile (128x64), convert fp32 -> bf16 through VALU
    for (int p = 0; p < 8; ++p) {
      int r = arow + p * 16;
      float4 vx = *reinterpret_cast<const float4*>(
          X + (size_t)(m0 + r) * E_ + k0 + acol);
      __bf16* d = &As[r * LDK + acol];
      d[0] = (__bf16)vx.x; d[1] = (__bf16)vx.y;
      d[2] = (__bf16)vx.z; d[3] = (__bf16)vx.w;
    }
    cp_wait();
    __syncthreads();

    for (int kk = 0; kk < 2; ++kk) {     // two 32-K WMMA slices
      v16bf af[4], bfv[2];
      for (int mt = 0; mt < 4; ++mt)
        af[mt] = ld16(&As[(wm + mt * 16 + fr) * LDK + kk * 32 + kh]);
      for (int nt = 0; nt < 2; ++nt)
        bfv[nt] = ld16(&Bs[(wn + nt * 16 + fr) * LDK + kk * 32 + kh]);
      for (int mt = 0; mt < 4; ++mt)
        for (int nt = 0; nt < 2; ++nt)
          acc[mt][nt] = wmma_bf16(af[mt], bfv[nt], acc[mt][nt]);
    }
    __syncthreads();
  }

  // epilogue: bias add, scatter into head-split layouts
  for (int mt = 0; mt < 4; ++mt) {
    for (int nt = 0; nt < 2; ++nt) {
      for (int r = 0; r < 8; ++r) {
        int gr = m0 + wm + mt * 16 + rbase + r;   // flat (b*S + s)
        int e  = n0 + wn + nt * 16 + fr;          // output feature
        float val = acc[mt][nt][r] + bias[e];
        int b = gr >> 11, s = gr & (S_ - 1);
        int h = e >> 6,  d = e & (DK_ - 1);
        if (z == 2) {
          Vt[(((size_t)b * H_ + h) * DK_ + d) * S_ + s] = (__bf16)val;
        } else {
          __bf16* dst = (z == 0) ? Qb : Kb;
          dst[(((size_t)b * H_ + h) * S_ + s) * DK_ + d] = (__bf16)val;
        }
      }
    }
  }
}

// ---------------- causal flash attention ----------------
// grid (S/128, H, B); 8 waves/block, each wave owns 16 query rows,
// key tile = 64 (amortizes softmax shuffles/exps over 2x WMMA work).
__global__ __launch_bounds__(256)
void attn_kernel(const __bf16* __restrict__ Q, const __bf16* __restrict__ K,
                 const __bf16* __restrict__ Vt, __bf16* __restrict__ O) {
  __shared__ __align__(32) __bf16 plds[8 * 16 * 64];  // per-wave P staging

  const int t = threadIdx.x, lane = t & 31, w = t >> 5;
  const int q0 = blockIdx.x * 128 + w * 16;
  const int h = blockIdx.y, b = blockIdx.z;
  const __bf16* Qh = Q  + (((size_t)b * H_ + h) * S_) * DK_;
  const __bf16* Kh = K  + (((size_t)b * H_ + h) * S_) * DK_;
  const __bf16* Vh = Vt + (((size_t)b * H_ + h) * DK_) * S_;
  const int fr = lane & 15, kh = (lane >> 4) * 16, rbase = (lane >> 4) * 8;
  __bf16* pl = &plds[w * 1024];

  // Q fragments (A layout): d = 0..31 and 32..63
  v16bf qa[2];
  qa[0] = ld16(Qh + (size_t)(q0 + fr) * DK_ + kh);
  qa[1] = ld16(Qh + (size_t)(q0 + fr) * DK_ + 32 + kh);

  float m_run[8], l_run[8];
  v8f o[4] = {};
  for (int r = 0; r < 8; ++r) { m_run[r] = -3.0e38f; l_run[r] = 0.f; }

  const float scale = 0.125f;  // 1/sqrt(DK)

  for (int j0 = 0; j0 <= q0 + 15; j0 += 64) {   // wave-uniform bound
    // speculative prefetch of the next key/value tiles (dropped if OOB)
    __builtin_prefetch(Kh + (size_t)(j0 + 64 + fr) * DK_, 0, 0);
    __builtin_prefetch(Vh + (size_t)fr * S_ + j0 + 64, 0, 0);

    // scores tile 16x64 = four 16x16 C frags, each summing two 32-K WMMAs
    v8f c[4];
    for (int sub = 0; sub < 4; ++sub) {
      v8f cc = {};
      for (int dc = 0; dc < 2; ++dc) {
        v16bf kb = ld16(Kh + (size_t)(j0 + sub * 16 + fr) * DK_ + dc * 32 + kh);
        cc = wmma_bf16(qa[dc], kb, cc);
      }
      c[sub] = cc;
    }
    // online softmax update (row r lives in VGPR r across a 16-lane half)
    for (int r = 0; r < 8; ++r) {
      int qi = q0 + rbase + r;
      float s[4];
      for (int sub = 0; sub < 4; ++sub) {
        s[sub] = c[sub][r] * scale;
        if (j0 + sub * 16 + fr > qi) s[sub] = -3.0e38f;   // causal mask
      }
      float mx = fmaxf(fmaxf(s[0], s[1]), fmaxf(s[2], s[3]));
      for (int d = 1; d < 16; d <<= 1) mx = fmaxf(mx, __shfl_xor(mx, d, 32));
      float m_new = fmaxf(m_run[r], mx);
      float p[4], rs = 0.f;
      for (int sub = 0; sub < 4; ++sub) {
        p[sub] = __expf(s[sub] - m_new);
        rs += p[sub];
      }
      for (int d = 1; d < 16; d <<= 1) rs += __shfl_xor(rs, d, 32);
      float corr = __expf(m_run[r] - m_new);
      l_run[r] = l_run[r] * corr + rs;
      m_run[r] = m_new;
      for (int dc = 0; dc < 4; ++dc) o[dc][r] *= corr;
      int M = rbase + r;
      for (int sub = 0; sub < 4; ++sub)
        pl[M * 64 + sub * 16 + fr] = (__bf16)p[sub];  // wave-private, DS in-order
    }
    // P as two A-fragments (16x32 each); V^T rows give contiguous B loads
    v16bf pa[2];
    pa[0] = ld16(&pl[fr * 64 + kh]);
    pa[1] = ld16(&pl[fr * 64 + 32 + kh]);
    for (int dc = 0; dc < 4; ++dc) {
      for (int kk = 0; kk < 2; ++kk) {
        v16bf vb = ld16(Vh + (size_t)(dc * 16 + fr) * S_ + j0 + kk * 32 + kh);
        o[dc] = wmma_bf16(pa[kk], vb, o[dc]);
      }
    }
  }

  // normalize and write [B,S,E] bf16
  for (int dc = 0; dc < 4; ++dc) {
    for (int r = 0; r < 8; ++r) {
      int s = q0 + rbase + r;
      int e = h * DK_ + dc * 16 + fr;
      float val = o[dc][r] / l_run[r];
      O[((size_t)b * S_ + s) * E_ + e] = (__bf16)val;
    }
  }
}

// ---------------- output projection: out = A @ Wo^T + bo (fp32 out) -------
__global__ __launch_bounds__(256)
void gemm_out_kernel(const __bf16* __restrict__ A, const __bf16* __restrict__ W,
                     const float* __restrict__ bias, float* __restrict__ out) {
  __shared__ __align__(32) __bf16 As[128 * LDK];
  __shared__ __align__(32) __bf16 Bs[128 * LDK];

  const int n0 = blockIdx.x * 128, m0 = blockIdx.y * 128;
  const int t = threadIdx.x, lane = t & 31, wid = t >> 5;
  const int wm = (wid & 1) * 64, wn = (wid >> 1) * 32;
  const int fr = lane & 15, kh = (lane >> 4) * 16, rbase = (lane >> 4) * 8;
  const int brow = t >> 3, bcol = (t & 7) * 8;

  v8f acc[4][2] = {};

  for (int k0 = 0; k0 < E_; k0 += TK) {
    // both tiles bf16: async DMA straight into LDS
    for (int p = 0; p < 4; ++p) {
      int r = brow + p * 32;
      cp16_g2l(A + (size_t)(m0 + r) * E_ + k0 + bcol, &As[r * LDK + bcol]);
      cp16_g2l(W + (size_t)(n0 + r) * E_ + k0 + bcol, &Bs[r * LDK + bcol]);
    }
    cp_wait();
    __syncthreads();

    for (int kk = 0; kk < 2; ++kk) {
      v16bf af[4], bfv[2];
      for (int mt = 0; mt < 4; ++mt)
        af[mt] = ld16(&As[(wm + mt * 16 + fr) * LDK + kk * 32 + kh]);
      for (int nt = 0; nt < 2; ++nt)
        bfv[nt] = ld16(&Bs[(wn + nt * 16 + fr) * LDK + kk * 32 + kh]);
      for (int mt = 0; mt < 4; ++mt)
        for (int nt = 0; nt < 2; ++nt)
          acc[mt][nt] = wmma_bf16(af[mt], bfv[nt], acc[mt][nt]);
    }
    __syncthreads();
  }

  for (int mt = 0; mt < 4; ++mt)
    for (int nt = 0; nt < 2; ++nt)
      for (int r = 0; r < 8; ++r) {
        int gr = m0 + wm + mt * 16 + rbase + r;
        int e  = n0 + wn + nt * 16 + fr;
        out[(size_t)gr * E_ + e] = acc[mt][nt][r] + bias[e];
      }
}

extern "C" void kernel_launch(void* const* d_in, const int* in_sizes, int n_in,
                              void* d_out, int out_size, void* d_ws, size_t ws_size,
                              hipStream_t stream) {
  const float* q   = (const float*)d_in[0];
  const float* k   = (const float*)d_in[1];
  const float* v   = (const float*)d_in[2];
  // d_in[3] = mask: causal tril, handled structurally in attn_kernel
  const float* wq  = (const float*)d_in[4];
  const float* bq  = (const float*)d_in[5];
  const float* wk  = (const float*)d_in[6];
  const float* bk  = (const float*)d_in[7];
  const float* wv  = (const float*)d_in[8];
  const float* bv  = (const float*)d_in[9];
  const float* wo  = (const float*)d_in[10];
  const float* bo  = (const float*)d_in[11];

  char* ws = (char*)d_ws;
  const size_t MB = 1024 * 1024;
  __bf16* wbf = (__bf16*)(ws);             //  8 MB: wq|wk|wv|wo bf16
  __bf16* Qb  = (__bf16*)(ws +  8 * MB);   //  8 MB: [B,H,S,DK]
  __bf16* Kb  = (__bf16*)(ws + 16 * MB);   //  8 MB: [B,H,S,DK]
  __bf16* Vt  = (__bf16*)(ws + 24 * MB);   //  8 MB: [B,H,DK,S]
  __bf16* AO  = (__bf16*)(ws + 32 * MB);   //  8 MB: [B,S,E]

  cvt_w_kernel<<<dim3(4 * E_ * E_ / 256), 256, 0, stream>>>(wq, wk, wv, wo, wbf);
  gemm_qkv_kernel<<<dim3(E_ / 128, (B_ * S_) / 128, 3), 256, 0, stream>>>(
      q, k, v, wbf, bq, bk, bv, Qb, Kb, Vt);
  attn_kernel<<<dim3(S_ / 128, H_, B_), 256, 0, stream>>>(Qb, Kb, Vt, AO);
  gemm_out_kernel<<<dim3(E_ / 128, (B_ * S_) / 128, 1), 256, 0, stream>>>(
      AO, wbf + (size_t)3 * E_ * E_, bo, (float*)d_out);
}
